// FlashRWAttention_21861383536740
// MI455X (gfx1250) — matrix operand order repaired and checked by reading, compiled, not run on hardware
//
#include <hip/hip_runtime.h>
#include <hip/hip_bf16.h>

// Problem constants (match the reference).
constexpr int kB  = 4;
constexpr int kS  = 1024;
constexpr int kH  = 4096;
constexpr int kNH = 32;
constexpr int kHD = 128;
constexpr int kT  = kB * kS;             // 4096 tokens
constexpr int kQKVC = kHD * (kNH + 2);   // 4352 qkv columns
constexpr int kD  = kNH * kHD;           // 4096 model dim
constexpr int kK  = kH;                  // compile-time GEMM depth (both GEMMs)

typedef __attribute__((ext_vector_type(16))) __bf16 v16bf;
typedef __attribute__((ext_vector_type(8)))  float  v8f;

union BF16Frag {
  v16bf v;
  uint4 q[2];
  __bf16 e[16];
};

__device__ __forceinline__ v8f wmma_bf16(const BF16Frag& a, const BF16Frag& b, v8f c) {
  // D = A(16x32 bf16) * B(32x16 bf16) + C(16x16 f32)
  return __builtin_amdgcn_wmma_f32_16x16x32_bf16(false, a.v, false, b.v,
                                                 (short)0, c, false, false);
}

// Async global -> LDS 16B copy (CDNA5 ASYNCcnt-tracked DMA path).
__device__ __forceinline__ void async_copy16(unsigned lds_off, const __bf16* gptr) {
  asm volatile("global_load_async_to_lds_b128 %0, %1, off"
               :: "v"(lds_off), "v"(gptr) : "memory");
}

// ---------------------------------------------------------------------------
// fp32 -> bf16 streaming convert
// ---------------------------------------------------------------------------
__global__ __launch_bounds__(256) void cvt_bf16_kernel(const float* __restrict__ src,
                                                       __bf16* __restrict__ dst,
                                                       size_t n) {
  size_t i = (size_t)blockIdx.x * blockDim.x + threadIdx.x;
  const size_t stride = (size_t)gridDim.x * blockDim.x;
  for (; i < n; i += stride) dst[i] = (__bf16)src[i];
}

// dst[C][R] = (bf16) src[R][C]   (pre-transposed weights for contiguous B-frags)
__global__ __launch_bounds__(256) void transpose_cvt_kernel(const float* __restrict__ src,
                                                            __bf16* __restrict__ dst,
                                                            int R, int C) {
  const size_t n = (size_t)R * C;
  size_t i = (size_t)blockIdx.x * blockDim.x + threadIdx.x;
  const size_t stride = (size_t)gridDim.x * blockDim.x;
  for (; i < n; i += stride) {
    const size_t c = i / (size_t)R;
    const size_t r = i - c * (size_t)R;
    dst[i] = (__bf16)src[r * (size_t)C + c];
  }
}

// ---------------------------------------------------------------------------
// C[M][N] (f32) = A[M][kK] (bf16, row-major) * BT[N][kK]^T (bf16)
// 8 waves per block; each wave owns a 16x64 C tile; block owns 128x64.
// B tile staged in LDS via async global->LDS DMA, double buffered; A frags
// double buffered in registers; B frags read into distinct registers so the
// four WMMAs per chunk issue back-to-back behind a single ds clause.
// ---------------------------------------------------------------------------
__global__ __launch_bounds__(256) void gemm_bf16_kernel(const __bf16* __restrict__ A,
                                                        const __bf16* __restrict__ BT,
                                                        float* __restrict__ C,
                                                        int N) {
  __shared__ alignas(16) __bf16 bbuf[2][64][32];   // 2 x 4KB double buffer
  const int tid = threadIdx.x;
  const int lane = tid & 31;
  const int wave = tid >> 5;
  const int lo = lane & 15;
  const int hi = lane >> 4;
  const int mBase = blockIdx.y * 128 + wave * 16;
  const int nBase = blockIdx.x * 64;

  // Cooperative async B staging: thread -> (row, 16B segment)
  const int brow = tid >> 2;          // 0..63
  const int bseg = (tid & 3) * 8;     // element offset 0/8/16/24
  const __bf16* bsrc = BT + (size_t)(nBase + brow) * kK + bseg;
  const unsigned lds0 = (unsigned)(uintptr_t)&bbuf[0][brow][bseg];
  const unsigned lds1 = (unsigned)(uintptr_t)&bbuf[1][brow][bseg];

  const __bf16* Ar = A + (size_t)(mBase + lo) * kK;

  v8f acc[4] = {};
  BF16Frag a0, a1;

  auto loadA = [&](BF16Frag& f, int k) {
    // A frag: lane holds M=lo; K = (h&7) + 8*hi + 16*(h>>3) -> two 16B loads
    f.q[0] = *(const uint4*)(Ar + k + 8 * hi);
    f.q[1] = *(const uint4*)(Ar + k + 16 + 8 * hi);
  };
  auto compute = [&](int buf, const BF16Frag& af) {
    // Load all four B fragments first (8 x ds_load_b128 as one clause),
    // then chain the four WMMAs back-to-back.
    BF16Frag bfr[4];
#pragma unroll
    for (int nt = 0; nt < 4; ++nt) {
      const __bf16* bp = &bbuf[buf][nt * 16 + lo][8 * hi];
      bfr[nt].q[0] = *(const uint4*)bp;
      bfr[nt].q[1] = *(const uint4*)(bp + 16);
    }
#pragma unroll
    for (int nt = 0; nt < 4; ++nt) acc[nt] = wmma_bf16(af, bfr[nt], acc[nt]);
  };

  // Software pipeline: issue(next) -> wait(cur) -> barrier -> compute(cur) -> barrier
  async_copy16(lds0, bsrc + 0);
  loadA(a0, 0);
#pragma unroll 1
  for (int k = 0; k < kK - 64; k += 64) {
    async_copy16(lds1, bsrc + k + 32);
    loadA(a1, k + 32);
    __builtin_prefetch(Ar + k + 256, 0, 1);
    asm volatile("s_wait_asynccnt 0x1" ::: "memory");
    __syncthreads();
    compute(0, a0);
    __syncthreads();
    async_copy16(lds0, bsrc + k + 64);
    loadA(a0, k + 64);
    asm volatile("s_wait_asynccnt 0x1" ::: "memory");
    __syncthreads();
    compute(1, a1);
    __syncthreads();
  }
  // Epilogue: last two chunks (kK-64 in buf0 already in flight, kK-32 in buf1)
  async_copy16(lds1, bsrc + kK - 32);
  loadA(a1, kK - 32);
  asm volatile("s_wait_asynccnt 0x1" ::: "memory");
  __syncthreads();
  compute(0, a0);
  __syncthreads();
  asm volatile("s_wait_asynccnt 0x0" ::: "memory");
  __syncthreads();
  compute(1, a1);

#pragma unroll
  for (int nt = 0; nt < 4; ++nt)
#pragma unroll
    for (int j = 0; j < 8; ++j)
      C[(size_t)(mBase + j + 8 * hi) * N + nBase + nt * 16 + lo] = acc[nt][j];
}

// ---------------------------------------------------------------------------
// RoPE + scatter: qkv f32 [T][4352] -> Qb bf16 [T][NH][HD],
// Kb bf16 [B][S][HD], Vt bf16 [B][HD][S] (transposed for P*V B-fragments)
// ---------------------------------------------------------------------------
__global__ __launch_bounds__(256) void rotary_scatter_kernel(const float* __restrict__ qkv,
                                                             const float* __restrict__ cosp,
                                                             const float* __restrict__ sinp,
                                                             __bf16* __restrict__ Qb,
                                                             __bf16* __restrict__ Kb,
                                                             __bf16* __restrict__ Vt) {
  const int t = blockIdx.x;
  const int b = t / kS;
  const int s = t - b * kS;
  const float* row = qkv + (size_t)t * kQKVC;
  const float* cs = cosp + (size_t)t * 64;
  const float* sn = sinp + (size_t)t * 64;

  for (int i = threadIdx.x; i < kNH * 64; i += blockDim.x) {
    const int h = i >> 6, d = i & 63;
    const float x1 = row[h * kHD + d];
    const float x2 = row[h * kHD + d + 64];
    const float c = cs[d], sv = sn[d];
    __bf16* qo = Qb + ((size_t)t * kNH + h) * kHD;
    qo[d]      = (__bf16)(x1 * c - x2 * sv);
    qo[d + 64] = (__bf16)(x2 * c + x1 * sv);
  }
  for (int d = threadIdx.x; d < 64; d += blockDim.x) {
    const float x1 = row[kNH * kHD + d];
    const float x2 = row[kNH * kHD + d + 64];
    const float c = cs[d], sv = sn[d];
    __bf16* ko = Kb + (size_t)t * kHD;
    ko[d]      = (__bf16)(x1 * c - x2 * sv);
    ko[d + 64] = (__bf16)(x2 * c + x1 * sv);
  }
  for (int d = threadIdx.x; d < kHD; d += blockDim.x)
    Vt[((size_t)b * kHD + d) * kS + s] = (__bf16)row[kNH * kHD + kHD + d];
}

// ---------------------------------------------------------------------------
// Causal MQA flash attention. One wave per 16-row Q tile, 4 waves per block.
// kv stepped by 32; online softmax; V B-fragments hoisted ahead of softmax so
// their load latency hides under the exp/reduce VALU work.
// ---------------------------------------------------------------------------
__global__ __launch_bounds__(128) void flash_attn_kernel(const __bf16* __restrict__ Q,
                                                         const __bf16* __restrict__ Kc,
                                                         const __bf16* __restrict__ Vt,
                                                         __bf16* __restrict__ O) {
  __shared__ alignas(16) __bf16 p_lds[4][16][32];
  const int lane = threadIdx.x & 31;
  const int wave = threadIdx.x >> 5;
  const int lo = lane & 15;
  const int hi = lane >> 4;
  const int blk = blockIdx.x;                 // B*NH*(S/64) blocks
  const int qt = blk & (kS / 64 - 1);
  const int h  = (blk / (kS / 64)) & (kNH - 1);
  const int b  = blk / ((kS / 64) * kNH);
  const int qBase = qt * 64 + wave * 16;

  // Q fragments (16x128 bf16, resident for whole kv loop)
  BF16Frag qf[4];
  const __bf16* qr = Q + ((size_t)(b * kS + qBase + lo) * kNH + h) * kHD;
#pragma unroll
  for (int c = 0; c < 4; ++c) {
    qf[c].q[0] = *(const uint4*)(qr + c * 32 + 8 * hi);
    qf[c].q[1] = *(const uint4*)(qr + c * 32 + 16 + 8 * hi);
  }

  v8f acc[8] = {};
  float m_i[8], l_i[8];
#pragma unroll
  for (int j = 0; j < 8; ++j) { m_i[j] = -1e30f; l_i[j] = 0.f; }
  const float scale = 0.08838834764831845f;   // 1/sqrt(128)
  const __bf16* Kb = Kc + (size_t)b * kS * kHD;
  const __bf16* Vb = Vt + (size_t)b * kHD * kS;
  const int kvEnd = qBase + 16;               // causal bound

  for (int kBase = 0; kBase < kvEnd; kBase += 32) {
    // Load all K fragments for both 16-col tiles first, then chain 8 WMMAs.
    BF16Frag kf0[4], kf1[4];
#pragma unroll
    for (int c = 0; c < 4; ++c) {
      const __bf16* kr = Kb + (size_t)(kBase + lo) * kHD + c * 32 + 8 * hi;
      kf0[c].q[0] = *(const uint4*)kr;
      kf0[c].q[1] = *(const uint4*)(kr + 16);
      const __bf16* kr2 = kr + 16 * kHD;
      kf1[c].q[0] = *(const uint4*)kr2;
      kf1[c].q[1] = *(const uint4*)(kr2 + 16);
    }
    v8f s0 = {}, s1 = {};
#pragma unroll
    for (int c = 0; c < 4; ++c) s0 = wmma_bf16(qf[c], kf0[c], s0);
#pragma unroll
    for (int c = 0; c < 4; ++c) s1 = wmma_bf16(qf[c], kf1[c], s1);

    // Hoist V B-fragments (independent of scores) so loads overlap softmax.
    BF16Frag vf[8];
#pragma unroll
    for (int dc = 0; dc < 8; ++dc) {
      const __bf16* vr = Vb + (size_t)(dc * 16 + lo) * kS + kBase + 8 * hi;
      vf[dc].q[0] = *(const uint4*)vr;
      vf[dc].q[1] = *(const uint4*)(vr + 16);
    }

    // scale + causal mask + online softmax (rows split j=0..7, M = j + 8*hi)
    float a0[8], a1[8], alpha[8];
#pragma unroll
    for (int j = 0; j < 8; ++j) {
      const int q = qBase + j + 8 * hi;
      float v0 = s0[j] * scale;
      float v1 = s1[j] * scale;
      if (kBase + lo > q)      v0 = -1e30f;
      if (kBase + 16 + lo > q) v1 = -1e30f;
      float mx = fmaxf(v0, v1);
#pragma unroll
      for (int off = 8; off > 0; off >>= 1) mx = fmaxf(mx, __shfl_xor(mx, off, 32));
      const float newm = fmaxf(m_i[j], mx);
      const float al = __expf(m_i[j] - newm);
      const float e0 = __expf(v0 - newm);
      const float e1 = __expf(v1 - newm);
      float rs = e0 + e1;
#pragma unroll
      for (int off = 8; off > 0; off >>= 1) rs += __shfl_xor(rs, off, 32);
      m_i[j] = newm;
      l_i[j] = l_i[j] * al + rs;
      alpha[j] = al;
      a0[j] = e0;
      a1[j] = e1;
    }
#pragma unroll
    for (int dc = 0; dc < 8; ++dc)
#pragma unroll
      for (int j = 0; j < 8; ++j) acc[dc][j] *= alpha[j];

    // P: D-layout -> A-layout through per-wave LDS tile (wave-local, no barrier)
#pragma unroll
    for (int j = 0; j < 8; ++j) {
      p_lds[wave][j + 8 * hi][lo]      = (__bf16)a0[j];
      p_lds[wave][j + 8 * hi][16 + lo] = (__bf16)a1[j];
    }
    asm volatile("s_wait_dscnt 0" ::: "memory");
    BF16Frag pf;
    const __bf16* pr = &p_lds[wave][lo][0];
    pf.q[0] = *(const uint4*)(pr + 8 * hi);
    pf.q[1] = *(const uint4*)(pr + 16 + 8 * hi);

    // O += P(16x32) * V(32x128)
#pragma unroll
    for (int dc = 0; dc < 8; ++dc) acc[dc] = wmma_bf16(pf, vf[dc], acc[dc]);
  }

  // normalize and emit bf16 [T][NH*HD] for the dense GEMM
#pragma unroll
  for (int dc = 0; dc < 8; ++dc) {
#pragma unroll
    for (int j = 0; j < 8; ++j) {
      const size_t t = (size_t)(b * kS + qBase + j + 8 * hi);
      O[t * kD + h * kHD + dc * 16 + lo] = (__bf16)(acc[dc][j] / l_i[j]);
    }
  }
}

// ---------------------------------------------------------------------------
extern "C" void kernel_launch(void* const* d_in, const int* in_sizes, int n_in,
                              void* d_out, int out_size, void* d_ws, size_t ws_size,
                              hipStream_t stream) {
  (void)in_sizes; (void)n_in; (void)out_size; (void)ws_size;
  const float* hs     = (const float*)d_in[0];  // [T][H]
  const float* wqkv   = (const float*)d_in[1];  // [H][4352]
  const float* wdense = (const float*)d_in[2];  // [H][H]
  const float* cosp   = (const float*)d_in[3];  // [T][1][64]
  const float* sinp   = (const float*)d_in[4];  // [T][1][64]
  // d_in[5] cu_seqlens, d_in[6] max_s: shapes are static here.

  char* p = (char*)d_ws;
  auto carve = [&p](size_t bytes) -> char* {
    char* r = p;
    p += (bytes + 255) & ~(size_t)255;
    return r;
  };
  __bf16* hsb   = (__bf16*)carve((size_t)kT * kH * sizeof(__bf16));
  __bf16* wqkvT = (__bf16*)carve((size_t)kQKVC * kH * sizeof(__bf16));
  __bf16* wdT   = (__bf16*)carve((size_t)kH * kH * sizeof(__bf16));
  float*  qkvf  = (float*) carve((size_t)kT * kQKVC * sizeof(float));
  __bf16* qb    = (__bf16*)carve((size_t)kT * kNH * kHD * sizeof(__bf16));
  __bf16* kb    = (__bf16*)carve((size_t)kB * kS * kHD * sizeof(__bf16));
  __bf16* vtb   = (__bf16*)carve((size_t)kB * kHD * kS * sizeof(__bf16));
  __bf16* ao    = (__bf16*)carve((size_t)kT * kD * sizeof(__bf16));

  cvt_bf16_kernel<<<4096, 256, 0, stream>>>(hs, hsb, (size_t)kT * kH);
  transpose_cvt_kernel<<<4096, 256, 0, stream>>>(wqkv, wqkvT, kH, kQKVC);
  transpose_cvt_kernel<<<4096, 256, 0, stream>>>(wdense, wdT, kH, kH);

  gemm_bf16_kernel<<<dim3(kQKVC / 64, kT / 128), 256, 0, stream>>>(
      hsb, wqkvT, qkvf, kQKVC);

  rotary_scatter_kernel<<<kT, 256, 0, stream>>>(qkvf, cosp, sinp, qb, kb, vtb);

  flash_attn_kernel<<<kB * kNH * (kS / 64), 128, 0, stream>>>(qb, kb, vtb, ao);

  gemm_bf16_kernel<<<dim3(kD / 64, kT / 128), 256, 0, stream>>>(
      ao, wdT, (float*)d_out, kD);
}
